// TriangularAttention_79087527789121
// MI455X (gfx1250) — matrix operand order repaired and checked by reading, not compiled, not run on hardware
//
#include <hip/hip_runtime.h>
#include <hip/hip_bf16.h>
#include <math.h>

typedef __attribute__((ext_vector_type(16))) _Float16 v16h;
typedef __attribute__((ext_vector_type(8)))  float    v8f;

#define NN 256
#define DD 32
#define HH 4
#define CC 4

// dot of a normalized-x row (4 channels) with column u of a (4 x 128) weight in LDS
__device__ __forceinline__ float dot4(float4 a, const float* w, int u) {
    return a.x * w[u] + a.y * w[128 + u] + a.z * w[256 + u] + a.w * w[384 + u];
}

// ---------------- Kernel 1: global mean / rsigma over 256*256*4 elements ----
__global__ __launch_bounds__(1024) void stats_kernel(const float* __restrict__ x,
                                                     float* __restrict__ stats) {
    __shared__ float ssum[1024];
    __shared__ float ssq[1024];
    const int tid = threadIdx.x;
    float s = 0.f, q = 0.f;
    for (int it = tid; it < NN * NN * CC; it += 1024) {
        float v = x[it];
        s += v;
        q += v * v;
    }
    ssum[tid] = s; ssq[tid] = q;
    __syncthreads();
    for (int off = 512; off > 0; off >>= 1) {
        if (tid < off) { ssum[tid] += ssum[tid + off]; ssq[tid] += ssq[tid + off]; }
        __syncthreads();
    }
    if (tid == 0) {
        const float inv_n = 1.0f / (float)(NN * NN * CC);
        float mean = ssum[0] * inv_n;
        float var  = ssq[0] * inv_n - mean * mean;
        stats[0] = mean;
        stats[1] = rsqrtf(var + 1e-5f);
    }
}

// ---------------- Kernel 2: normalized x and bias tensor --------------------
__global__ __launch_bounds__(256) void xnbias_kernel(const float* __restrict__ x,
                                                     const float* __restrict__ bw, // (C,H)
                                                     const float* __restrict__ stats,
                                                     float* __restrict__ xn,
                                                     float* __restrict__ biasb) {
    const int idx = blockIdx.x * 256 + threadIdx.x; // flat (a,b) over N*N
    const float mu = stats[0], rs = stats[1];
    float c0 = (x[idx * 4 + 0] - mu) * rs;
    float c1 = (x[idx * 4 + 1] - mu) * rs;
    float c2 = (x[idx * 4 + 2] - mu) * rs;
    float c3 = (x[idx * 4 + 3] - mu) * rs;
    xn[idx * 4 + 0] = c0; xn[idx * 4 + 1] = c1;
    xn[idx * 4 + 2] = c2; xn[idx * 4 + 3] = c3;
#pragma unroll
    for (int h = 0; h < HH; ++h) {
        biasb[idx * 4 + h] =
            c0 * bw[0 * 4 + h] + c1 * bw[1 * 4 + h] + c2 * bw[2 * 4 + h] + c3 * bw[3 * 4 + h];
    }
}

// ---------------- Kernel 3: fused triangular attention per row i ------------
// grid = 256 (i), block = 256 threads = 8 waves; wave -> (head, j-half)
__global__ __launch_bounds__(256) void attn_kernel(
    const float* __restrict__ xn_g,    // (N*N, 4) normalized
    const float* __restrict__ bias_g,  // (N*N, H)
    const unsigned char* __restrict__ mask, // (N*N) bool
    const float* __restrict__ qw_g, const float* __restrict__ kw_g,
    const float* __restrict__ vw_g, const float* __restrict__ gw_g,
    const float* __restrict__ gb_g, const float* __restrict__ ow_g,
    const float* __restrict__ ob_g,
    float* __restrict__ out) {

    __shared__ float4 xns[NN];          // normalized x rows of this i-block
    __shared__ float sqw[512], skw[512], svw[512], sgw[512], sow[512];
    __shared__ float sgb[128];
    __shared__ float smask[NN];
    __shared__ float sob[4];
    __shared__ float out_head[HH][NN][CC];

    const int i   = blockIdx.x;
    const int tid = threadIdx.x;

    for (int u = tid; u < 512; u += 256) {
        sqw[u] = qw_g[u]; skw[u] = kw_g[u]; svw[u] = vw_g[u];
        sgw[u] = gw_g[u]; sow[u] = ow_g[u];
    }
    if (tid < 128) sgb[tid] = gb_g[tid];
    if (tid < 4)   sob[tid] = ob_g[tid];
    {
        const float4* xr = (const float4*)(xn_g + (size_t)i * NN * CC);
        xns[tid]   = xr[tid];
        smask[tid] = mask[i * NN + tid] ? 1.f : 0.f;
    }
    __syncthreads();

    const int  wave  = tid >> 5;
    const int  lane  = tid & 31;
    const int  h     = wave >> 1;   // head
    const int  jhalf = wave & 1;    // which 128 rows of j
    const int  n     = lane & 15;
    const bool hi    = lane >= 16;
    const float scale = 0.17677669529663687f; // 1/sqrt(32)

    for (int jb = 0; jb < 8; ++jb) {
        const int jbase = jhalf * 128 + jb * 16;
        const int jg    = jbase + n;           // this lane's j column
        const float4 xj = xns[jg];

        // ---- B-operand: Q^T (K=d, N=j). Lane n = j; lo lanes d=0..15, hi d=16..31.
        v16h qf;
#pragma unroll
        for (int r = 0; r < 8; ++r) {
            int d0 = 2 * r + (hi ? 16 : 0);
            qf[2 * r]     = (_Float16)dot4(xj, sqw, (d0)     * 4 + h);
            qf[2 * r + 1] = (_Float16)dot4(xj, sqw, (d0 + 1) * 4 + h);
        }

        // ---- S^T tiles: rows = k (16 per tile), cols = j; one WMMA each (K=D=32)
        v8f st[16];
#pragma unroll
        for (int t = 0; t < 16; ++t) {
            const int kg = 16 * t + n;         // A-matrix row = k
            const float4 xk = xns[kg];
            v16h kf;                           // A layout: per-lane d pattern
#pragma unroll
            for (int r = 0; r < 8; ++r) {
                int d0 = 2 * r + (hi ? 16 : 0);
                kf[2 * r]     = (_Float16)dot4(xk, skw, (d0)     * 4 + h);
                kf[2 * r + 1] = (_Float16)dot4(xk, skw, (d0 + 1) * 4 + h);
            }
            v8f z = {};
            st[t] = __builtin_amdgcn_wmma_f32_16x16x32_f16(false, kf, false, qf,
                                                           (short)0, z, false, false);
        }

        // ---- bias add + softmax over k (per-lane components + one cross-half xor)
        float mrow = -3.0e38f;
#pragma unroll
        for (int t = 0; t < 16; ++t) {
#pragma unroll
            for (int r = 0; r < 8; ++r) {
                int kg = 16 * t + r + (hi ? 8 : 0);
                float s = st[t][r] + bias_g[((size_t)jg * NN + kg) * HH + h];
                st[t][r] = s;
                mrow = fmaxf(mrow, s);
            }
        }
        mrow = fmaxf(mrow, __shfl_xor(mrow, 16, 32));
        float sum = 0.f;
#pragma unroll
        for (int t = 0; t < 16; ++t) {
#pragma unroll
            for (int r = 0; r < 8; ++r) {
                float e = __expf((st[t][r] - mrow) * scale);
                st[t][r] = e;
                sum += e;
            }
        }
        sum += __shfl_xor(sum, 16, 32);
        const float inv   = 1.f / sum;
        const float keep  = smask[jg];         // masked (i,j) row -> uniform 1/256
        const float uni   = 1.0f / 256.0f;
#pragma unroll
        for (int t = 0; t < 16; ++t) {
#pragma unroll
            for (int r = 0; r < 8; ++r)
                st[t][r] = (keep > 0.5f) ? st[t][r] * inv : uni;
        }

        // ---- O^T = V^T * P^T, accumulated over 8 k-steps of 32
        v8f oacc0 = {}, oacc1 = {};
#pragma unroll
        for (int ks = 0; ks < 8; ++ks) {
            // repack two C-layout 16x16 P tiles into B-operand P^T (K=k, N=j)
            float ownv[8], rcvv[8];
#pragma unroll
            for (int r = 0; r < 8; ++r) {
                float give = hi ? st[2 * ks][r] : st[2 * ks + 1][r];
                rcvv[r] = __shfl_xor(give, 16, 32);
                ownv[r] = hi ? st[2 * ks + 1][r] : st[2 * ks][r];
            }
            v16h pf;
#pragma unroll
            for (int q = 0; q < 8; ++q) {
                pf[q]     = (_Float16)(hi ? rcvv[q] : ownv[q]);
                pf[q + 8] = (_Float16)(hi ? ownv[q] : rcvv[q]);
            }
#pragma unroll
            for (int dt = 0; dt < 2; ++dt) {
                const int d = dt * 16 + n;     // A-matrix row = d
                v16h vf;                       // V^T, K = k (A-layout pattern)
#pragma unroll
                for (int r = 0; r < 8; ++r) {
                    int k0 = 2 * r + (hi ? 16 : 0);
                    k0 += 32 * ks;
                    vf[2 * r]     = (_Float16)dot4(xns[k0],     svw, d * 4 + h);
                    vf[2 * r + 1] = (_Float16)dot4(xns[k0 + 1], svw, d * 4 + h);
                }
                if (dt == 0)
                    oacc0 = __builtin_amdgcn_wmma_f32_16x16x32_f16(false, vf, false, pf,
                                                                   (short)0, oacc0, false, false);
                else
                    oacc1 = __builtin_amdgcn_wmma_f32_16x16x32_f16(false, vf, false, pf,
                                                                   (short)0, oacc1, false, false);
            }
        }

        // ---- fused gate + output projection (per-lane partials over this head's 32 d's)
        float pc[4] = {0.f, 0.f, 0.f, 0.f};
#pragma unroll
        for (int dt = 0; dt < 2; ++dt) {
#pragma unroll
            for (int r = 0; r < 8; ++r) {
                int d = dt * 16 + r + (hi ? 8 : 0);
                int u = d * 4 + h;
                float wo = (dt == 0) ? oacc0[r] : oacc1[r];
                float g  = 1.f / (1.f + __expf(-(dot4(xj, sgw, u) + sgb[u])));
                float val = wo * g;
#pragma unroll
                for (int c = 0; c < 4; ++c) pc[c] += val * sow[u * 4 + c];
            }
        }
#pragma unroll
        for (int c = 0; c < 4; ++c) pc[c] += __shfl_xor(pc[c], 16, 32);
        if (!hi) {
#pragma unroll
            for (int c = 0; c < 4; ++c) out_head[h][jg][c] = pc[c];
        }
    }

    __syncthreads();
    {   // final deterministic head reduction: one thread per j
        const int j = tid;
#pragma unroll
        for (int c = 0; c < 4; ++c) {
            float acc = sob[c];
#pragma unroll
            for (int hh = 0; hh < HH; ++hh) acc += out_head[hh][j][c];
            out[((size_t)i * NN + j) * CC + c] = acc;
        }
    }
}

// ---------------------------------------------------------------------------
extern "C" void kernel_launch(void* const* d_in, const int* in_sizes, int n_in,
                              void* d_out, int out_size, void* d_ws, size_t ws_size,
                              hipStream_t stream) {
    const float*         x      = (const float*)d_in[0];
    const unsigned char* mask   = (const unsigned char*)d_in[1];
    const float*         q_w    = (const float*)d_in[2];
    const float*         k_w    = (const float*)d_in[3];
    const float*         v_w    = (const float*)d_in[4];
    const float*         b_w    = (const float*)d_in[5];
    const float*         gate_w = (const float*)d_in[6];
    const float*         gate_b = (const float*)d_in[7];
    const float*         out_w  = (const float*)d_in[8];
    const float*         out_b  = (const float*)d_in[9];
    float* out = (float*)d_out;

    float* ws    = (float*)d_ws;
    float* stats = ws;                          // 2 floats (+pad)
    float* xn    = ws + 16;                     // 256*256*4 floats
    float* biasb = ws + 16 + NN * NN * CC;      // 256*256*4 floats

    stats_kernel<<<1, 1024, 0, stream>>>(x, stats);
    xnbias_kernel<<<NN, 256, 0, stream>>>(x, b_w, stats, xn, biasb);
    attn_kernel<<<NN, 256, 0, stream>>>(xn, biasb, mask, q_w, k_w, v_w,
                                        gate_w, gate_b, out_w, out_b, out);
    (void)in_sizes; (void)n_in; (void)out_size; (void)ws_size;
}
